// SelfAttention_27350351741347
// MI455X (gfx1250) — matrix operand order repaired and checked by reading, not compiled, not run on hardware
//
#include <hip/hip_runtime.h>

// Problem dims (fixed by the reference)
#define BATCH   2
#define S_LEN   2048
#define D_DIM   2048
#define NHEAD   16
#define HEAD_DIM 128

typedef __attribute__((ext_vector_type(16))) __bf16 v16bf;
typedef __attribute__((ext_vector_type(8)))  float  v8f;

__device__ __forceinline__ unsigned short f32_to_bf16_rne(float f) {
    union { float f; unsigned int u; } c; c.f = f;
    unsigned int u = c.u;
    u += 0x7FFFu + ((u >> 16) & 1u);     // round-to-nearest-even
    return (unsigned short)(u >> 16);
}

union AFrag { v16bf v; unsigned int u[8]; };

__device__ __forceinline__ v8f wmma_bf16(v16bf a, v16bf b, v8f c) {
    // D = A(16x32 bf16) * B(32x16 bf16) + C(16x16 f32)
    return __builtin_amdgcn_wmma_f32_16x16x32_bf16(
        /*neg_a=*/false, a, /*neg_b=*/false, b,
        /*c_mod=*/(short)0, c, /*reuse_a=*/false, /*reuse_b=*/false);
}

// ---------------------------------------------------------------------------
// Generic GEMM:  Out(M,N) = A(M,K) @ W(K,N) + bias
//   A: f32 (A_BF16=false) or bf16-as-ushort (A_BF16=true), row-major
//   W: f32 row-major, converted to bf16 in LDS
//   Out: bf16-as-ushort (OUT_BF16=true) or f32
// Block: 256 threads = 8 wave32 waves; 128x128 tile; wave tile 32x64.
// ---------------------------------------------------------------------------
template <bool A_BF16, bool OUT_BF16>
__global__ __launch_bounds__(256)
void gemm_bf16_wmma(const void* __restrict__ Aptr,
                    const float* __restrict__ W,
                    const float* __restrict__ bias,
                    void* __restrict__ Out,
                    int M, int K, int N) {
    __shared__ __align__(16) unsigned short sA[128][40];  // [m][k], +8 pad
    __shared__ __align__(16) unsigned short sB[128][40];  // [n][k], +8 pad

    const int rowBase = blockIdx.y * 128;
    const int colBase = blockIdx.x * 128;
    const int lane = threadIdx.x & 31;
    const int w    = threadIdx.x >> 5;       // wave 0..7
    const int g    = lane >> 4;              // half-wave 0/1
    const int mr   = lane & 15;
    const int wm   = (w >> 1) * 32;          // wave row offset in tile
    const int wn   = (w & 1) * 64;           // wave col offset in tile

    v8f acc[2][4];
#pragma unroll
    for (int tm = 0; tm < 2; ++tm)
#pragma unroll
        for (int tn = 0; tn < 4; ++tn) acc[tm][tn] = {};

    const int ksteps = K >> 5;               // K / 32
    for (int kt = 0; kt < ksteps; ++kt) {
        // ---- cooperative load A tile: 128 rows x 32 k -> bf16 LDS ----
#pragma unroll
        for (int r = 0; r < 4; ++r) {
            const int linear = threadIdx.x + 256 * r;    // 0..1023 quads
            const int row = linear >> 3;                 // /8 -> 0..127
            const int c4  = linear & 7;                  // quad col
            unsigned int p0, p1;
            if constexpr (A_BF16) {
                const unsigned short* A = (const unsigned short*)Aptr;
                uint2 d = *(const uint2*)(const void*)
                    (A + (size_t)(rowBase + row) * K + kt * 32 + c4 * 4);
                p0 = d.x; p1 = d.y;
            } else {
                const float* A = (const float*)Aptr;
                float4 f = *(const float4*)(const void*)
                    (A + (size_t)(rowBase + row) * K + kt * 32 + c4 * 4);
                p0 = (unsigned int)f32_to_bf16_rne(f.x) |
                     ((unsigned int)f32_to_bf16_rne(f.y) << 16);
                p1 = (unsigned int)f32_to_bf16_rne(f.z) |
                     ((unsigned int)f32_to_bf16_rne(f.w) << 16);
            }
            *(uint2*)(void*)&sA[row][c4 * 4] = make_uint2(p0, p1);
        }
        // ---- cooperative load W tile: 32 k x 128 n, store transposed [n][k] ----
#pragma unroll
        for (int r = 0; r < 4; ++r) {
            const int linear = threadIdx.x + 256 * r;    // 0..1023 quads
            const int k  = linear >> 5;                  // /32 -> 0..31
            const int n4 = linear & 31;
            float4 f = *(const float4*)(const void*)
                (W + (size_t)(kt * 32 + k) * N + colBase + n4 * 4);
            sB[n4 * 4 + 0][k] = f32_to_bf16_rne(f.x);
            sB[n4 * 4 + 1][k] = f32_to_bf16_rne(f.y);
            sB[n4 * 4 + 2][k] = f32_to_bf16_rne(f.z);
            sB[n4 * 4 + 3][k] = f32_to_bf16_rne(f.w);
        }
        __syncthreads();

        // ---- fragments + 8 WMMA ----
        AFrag af[2];
#pragma unroll
        for (int tm = 0; tm < 2; ++tm)
#pragma unroll
            for (int j = 0; j < 8; ++j) {
                const int k0 = 2 * j + 8 * g + ((j >= 4) ? 8 : 0);   // A 16x32 layout
                af[tm].u[j] = *(const unsigned int*)(const void*)
                              &sA[wm + tm * 16 + mr][k0];
            }
        AFrag bf[4];
#pragma unroll
        for (int tn = 0; tn < 4; ++tn)
#pragma unroll
            for (int j = 0; j < 8; ++j) {
                const int k0 = 2 * j + 8 * g + ((j >= 4) ? 8 : 0);   // B 32x16 layout (same k order as A)
                bf[tn].u[j] = *(const unsigned int*)(const void*)
                              &sB[wn + tn * 16 + mr][k0];
            }
#pragma unroll
        for (int tm = 0; tm < 2; ++tm)
#pragma unroll
            for (int tn = 0; tn < 4; ++tn)
                acc[tm][tn] = wmma_bf16(af[tm].v, bf[tn].v, acc[tm][tn]);
        __syncthreads();
    }

    // ---- epilogue: bias add + store (C layout: row = i + 8g, col = mr) ----
#pragma unroll
    for (int tm = 0; tm < 2; ++tm)
#pragma unroll
        for (int tn = 0; tn < 4; ++tn)
#pragma unroll
            for (int i = 0; i < 8; ++i) {
                const int row = rowBase + wm + tm * 16 + 8 * g + i;
                const int col = colBase + wn + tn * 16 + mr;
                const float v = acc[tm][tn][i] + bias[col];
                if constexpr (OUT_BF16)
                    ((unsigned short*)Out)[(size_t)row * N + col] = f32_to_bf16_rne(v);
                else
                    ((float*)Out)[(size_t)row * N + col] = v;
            }
}

// ---------------------------------------------------------------------------
// Flash attention w/ causal mask + ALiBi. Q/K/V bf16 laid out (B, S, H*HD),
// head base offset b*S*D + h*HD, row stride D. One workgroup = 128 q rows of
// one (b,h); each wave owns 16 q rows (independent online softmax).
// ---------------------------------------------------------------------------
__global__ __launch_bounds__(256)
void flash_attn_alibi(const unsigned short* __restrict__ Q,
                      const unsigned short* __restrict__ Kt,
                      const unsigned short* __restrict__ Vt,
                      unsigned short* __restrict__ Ctx) {
    __shared__ __align__(16) unsigned short sK[64][132];     // [kv][d]
    __shared__ __align__(16) unsigned short sV[128][68];     // [d][kv] (transposed)
    __shared__ __align__(16) unsigned short sP[8][16][68];   // per-wave P tile

    const int bh = blockIdx.y;
    const int b = bh / NHEAD, h = bh % NHEAD;
    const int qb = blockIdx.x * 128;
    const int w = threadIdx.x >> 5, lane = threadIdx.x & 31;
    const int g = lane >> 4, mr = lane & 15;
    const int q0 = qb + w * 16;                  // this wave's first q row

    const size_t headOff = (size_t)b * S_LEN * D_DIM + (size_t)h * HEAD_DIM;
    const unsigned short* Qh = Q + headOff;
    const unsigned short* Kh = Kt + headOff;
    const unsigned short* Vh = Vt + headOff;
    unsigned short*       Ch = Ctx + headOff;

    // Q A-fragments for the 4 k-chunks of HD=128 (loaded once, before Ctx writes)
    AFrag qf[4];
#pragma unroll
    for (int c = 0; c < 4; ++c)
#pragma unroll
        for (int j = 0; j < 8; ++j) {
            const int k0 = 2 * j + 8 * g + ((j >= 4) ? 8 : 0);
            qf[c].u[j] = *(const unsigned int*)(const void*)
                         (Qh + (size_t)(q0 + mr) * D_DIM + c * 32 + k0);
        }

    v8f o[8];
#pragma unroll
    for (int dt = 0; dt < 8; ++dt) o[dt] = {};
    float mstat[8], lstat[8];
#pragma unroll
    for (int i = 0; i < 8; ++i) { mstat[i] = -3.0e38f; lstat[i] = 0.f; }

    const float scale = 0.08838834764831845f;           // 1/sqrt(128)
    const float slope = exp2f(-0.5f * (float)(h + 1));  // ALiBi, H=16

    const int nblk = 2 * blockIdx.x + 2;                // causal: kv in [0, qb+128)
    for (int blk = 0; blk < nblk; ++blk) {
        const int kv0 = blk * 64;
        // ---- cooperative stage: K tile 64x128, V tile transposed ----
#pragma unroll
        for (int r = 0; r < 8; ++r) {
            const int linear = threadIdx.x + 256 * r;   // 0..2047 quads
            const int row = linear >> 5;                // kv 0..63
            const int c4  = linear & 31;
            uint2 dk = *(const uint2*)(const void*)
                (Kh + (size_t)(kv0 + row) * D_DIM + c4 * 4);
            *(uint2*)(void*)&sK[row][c4 * 4] = dk;
            uint2 dv = *(const uint2*)(const void*)
                (Vh + (size_t)(kv0 + row) * D_DIM + c4 * 4);
            sV[c4 * 4 + 0][row] = (unsigned short)(dv.x & 0xffffu);
            sV[c4 * 4 + 1][row] = (unsigned short)(dv.x >> 16);
            sV[c4 * 4 + 2][row] = (unsigned short)(dv.y & 0xffffu);
            sV[c4 * 4 + 3][row] = (unsigned short)(dv.y >> 16);
        }
        __syncthreads();

        if (kv0 <= q0 + 15) {   // wave-uniform: block not fully masked for this wave
            // ---- scores: S(16 x 64) = Q(16x128) @ K^T, 4 tiles x 4 k-chunks ----
            v8f st[4];
#pragma unroll
            for (int t = 0; t < 4; ++t) {
                v8f a = {};
#pragma unroll
                for (int c = 0; c < 4; ++c) {
                    AFrag bf;
#pragma unroll
                    for (int j = 0; j < 8; ++j) {
                        const int k0 = 2 * j + 8 * g + ((j >= 4) ? 8 : 0);
                        bf.u[j] = *(const unsigned int*)(const void*)
                                  &sK[t * 16 + mr][c * 32 + k0];
                    }
                    a = wmma_bf16(qf[c].v, bf.v, a);
                }
                st[t] = a;
            }
            // ---- online softmax (row = q0 + 8g + i, col = kv0 + 16t + mr) ----
#pragma unroll
            for (int i = 0; i < 8; ++i) {
                const int qrow = q0 + 8 * g + i;
                float mx = -3.0e38f;
#pragma unroll
                for (int t = 0; t < 4; ++t) {
                    const int kvc = kv0 + t * 16 + mr;
                    float v = st[t][i] * scale + slope * (float)(kvc - (S_LEN - 1));
                    if (kvc > qrow) v = -3.0e38f;       // causal mask
                    st[t][i] = v;
                    mx = fmaxf(mx, v);
                }
#pragma unroll
                for (int m = 1; m < 16; m <<= 1) mx = fmaxf(mx, __shfl_xor(mx, m, 32));
                const float mnew  = fmaxf(mstat[i], mx);
                const float alpha = __expf(mstat[i] - mnew);
                float rs = 0.f;
#pragma unroll
                for (int t = 0; t < 4; ++t) {
                    const float p = __expf(st[t][i] - mnew);
                    st[t][i] = p;
                    rs += p;
                }
#pragma unroll
                for (int m = 1; m < 16; m <<= 1) rs += __shfl_xor(rs, m, 32);
                lstat[i] = lstat[i] * alpha + rs;
                mstat[i] = mnew;
#pragma unroll
                for (int t = 0; t < 4; ++t)
                    sP[w][8 * g + i][t * 16 + mr] = f32_to_bf16_rne(st[t][i]);
#pragma unroll
                for (int dt = 0; dt < 8; ++dt) o[dt][i] *= alpha;
            }
            // ---- O += P(16x64) @ V(64x128): 2 kv chunks x 8 d tiles ----
#pragma unroll
            for (int c = 0; c < 2; ++c) {
                AFrag pf;
#pragma unroll
                for (int j = 0; j < 8; ++j) {
                    const int k0 = 2 * j + 8 * g + ((j >= 4) ? 8 : 0);
                    pf.u[j] = *(const unsigned int*)(const void*)
                              &sP[w][mr][c * 32 + k0];
                }
#pragma unroll
                for (int dt = 0; dt < 8; ++dt) {
                    AFrag bf;
#pragma unroll
                    for (int j = 0; j < 8; ++j) {
                        const int k0 = 2 * j + 8 * g + ((j >= 4) ? 8 : 0);
                        bf.u[j] = *(const unsigned int*)(const void*)
                                  &sV[dt * 16 + mr][c * 32 + k0];
                    }
                    o[dt] = wmma_bf16(pf.v, bf.v, o[dt]);
                }
            }
        }
        __syncthreads();
    }

    // ---- finalize: ctx = O / l, store bf16 (same rows this WG read as Q) ----
#pragma unroll
    for (int dt = 0; dt < 8; ++dt)
#pragma unroll
        for (int i = 0; i < 8; ++i) {
            const int qrow = q0 + 8 * g + i;
            const int dcol = dt * 16 + mr;
            const float v = o[dt][i] / lstat[i];
            Ch[(size_t)qrow * D_DIM + dcol] = f32_to_bf16_rne(v);
        }
}

// ---------------------------------------------------------------------------
extern "C" void kernel_launch(void* const* d_in, const int* in_sizes, int n_in,
                              void* d_out, int out_size, void* d_ws, size_t ws_size,
                              hipStream_t stream) {
    const float* x  = (const float*)d_in[0];
    const float* Wq = (const float*)d_in[1];
    const float* Wk = (const float*)d_in[2];
    const float* Wv = (const float*)d_in[3];
    const float* Wo = (const float*)d_in[4];
    const float* bq = (const float*)d_in[5];
    const float* bk = (const float*)d_in[6];
    const float* bv = (const float*)d_in[7];
    const float* bo = (const float*)d_in[8];
    float* out = (float*)d_out;

    const size_t elems = (size_t)BATCH * S_LEN * D_DIM;   // 8M bf16 = 16 MB each
    unsigned short* qws = (unsigned short*)d_ws;
    unsigned short* kws = qws + elems;
    unsigned short* vws = kws + elems;
    // ctx reuses qws: each attention workgroup reads its Q rows into registers
    // before writing ctx to the same (disjoint-per-workgroup) region.

    const int M = BATCH * S_LEN, K = D_DIM, N = D_DIM;
    dim3 blk(256);
    dim3 gG(N / 128, M / 128);

    gemm_bf16_wmma<false, true><<<gG, blk, 0, stream>>>(x, Wq, bq, qws, M, K, N);
    gemm_bf16_wmma<false, true><<<gG, blk, 0, stream>>>(x, Wk, bk, kws, M, K, N);
    gemm_bf16_wmma<false, true><<<gG, blk, 0, stream>>>(x, Wv, bv, vws, M, K, N);

    flash_attn_alibi<<<dim3(S_LEN / 128, BATCH * NHEAD), blk, 0, stream>>>(
        qws, kws, vws, qws);

    gemm_bf16_wmma<true, false><<<gG, blk, 0, stream>>>(qws, Wo, bo, out, M, K, N);
}
